// RPN_87703232185028
// MI455X (gfx1250) — compile-verified, matrix operand order checked by reading, not run on hardware
//
#include <hip/hip_runtime.h>
#include <stdint.h>
#include <math.h>

typedef __attribute__((ext_vector_type(16))) _Float16 v16h;
typedef __attribute__((ext_vector_type(8)))  _Float16 v8h;
typedef __attribute__((ext_vector_type(8)))  float    v8f;

#define FEAT     96
#define NPIX     (FEAT*FEAT)          // 9216
#define C_IN     1024
#define C_MID    512
#define KANCH    9
#define N_ANCH   (NPIX*KANCH)         // 82944
#define KDIM     (C_IN*9)             // 9216
#define PRE_TOP  6000
#define POS_TOP  300
#define NMS_THR_ 0.7f
#define N_SORT   131072               // 2^17 >= 82944
#define NBLK64   ((PRE_TOP+63)/64)    // 94
#define IMG_MAX  1535.0f
#define PH       98                   // padded image height
#define PW       128                  // padded image row stride (256B)
#define PLANE    ((size_t)C_IN*PH*PW) // halves per kx-plane

// ---------------- CDNA5 async global->LDS helpers ----------------
__device__ __forceinline__ void async_ld_b128(unsigned lds_off, const void* gaddr) {
    asm volatile("global_load_async_to_lds_b128 %0, %1, off"
                 :: "v"(lds_off), "v"(gaddr) : "memory");
}
__device__ __forceinline__ void wait_async0() {
    asm volatile("s_wait_asynccnt 0x0" ::: "memory");
}

// ---------------- fragment helper (16-bit A/B 16x32 lane layout) ----------
// lane L: row/col = L&15, g = L>>4. halves 0..7 -> K = g*8+0..7,
// halves 8..15 -> K = 16+g*8+0..7. Fragment = two 16B LDS reads.
__device__ __forceinline__ v16h load_frag(const _Float16* row, int g) {
    union { v16h v; v8h h[2]; } u;
    u.h[0] = *(const v8h*)(row + g*8);
    u.h[1] = *(const v8h*)(row + 16 + g*8);
    return u.v;
}

// ---------------- pack kernels ----------------
// x[1024][96][96] f32 -> 3 kx-shifted zero-padded f16 planes Xp3[3][1024][98][128]
__global__ void pack_x_shift3(const float* __restrict__ x, _Float16* __restrict__ Xp3, int n) {
    int i = blockIdx.x*blockDim.x + threadIdx.x;
    if (i >= n) return;
    int kx  = i / (int)PLANE;
    int rem = i - kx * (int)PLANE;
    int ci = rem / (PH*PW), r = (rem / PW) % PH, c = rem % PW;
    int yy = r - 1, xx = c - 1 + kx;
    float v = 0.0f;
    if (yy >= 0 && yy < FEAT && xx >= 0 && xx < FEAT)
        v = x[(size_t)ci*NPIX + yy*FEAT + xx];
    Xp3[i] = (_Float16)v;
}

// w_rpn [co][ci][ky][kx] -> Wp[co][k], k = (ky*3+kx)*1024 + ci (k contiguous)
__global__ void pack_w_rpn(const float* __restrict__ w, _Float16* __restrict__ Wp) {
    int i = blockIdx.x*blockDim.x + threadIdx.x;
    if (i >= C_MID*KDIM) return;
    int co = i / KDIM, k = i - co*KDIM;
    int khw = k >> 10, ci = k & 1023;
    Wp[i] = (_Float16)w[(size_t)co*KDIM + ci*9 + khw];
}

// cls(18x512) + reg(36x512) -> Wcr[64][512] (rows 54..63 zero), k contiguous
__global__ void pack_wcr(const float* __restrict__ wc, const float* __restrict__ wr,
                         _Float16* __restrict__ Wcr) {
    int i = blockIdx.x*blockDim.x + threadIdx.x;
    if (i >= 64*C_MID) return;
    int o = i / C_MID, ci = i - o*C_MID;
    float v = 0.0f;
    if (o < 18)      v = wc[o*C_MID + ci];
    else if (o < 54) v = wr[(o-18)*C_MID + ci];
    Wcr[i] = (_Float16)v;
}

// ---------------- conv 3x3 as implicit-GEMM WMMA (async-A, vector-B) -----
// Hh[p][co] = relu( sum_k Wp[co][k] * im2col(Xp3)[k][p] + b[co] )
__global__ __launch_bounds__(256)
void conv3x3_wmma(const _Float16* __restrict__ Xp3,  // [3][C_IN][98][128]
                  const _Float16* __restrict__ Wp,   // [C_MID][KDIM]
                  const float* __restrict__ brpn,
                  _Float16* __restrict__ Hh)         // [NPIX][C_MID]
{
    __shared__ __align__(16) _Float16 sA[2][128*40]; // [m][k], dbl-buffered (async)
    __shared__ __align__(16) _Float16 sB[128*40];    // [n][k]
    const int t = threadIdx.x;
    const int wave = t >> 5, lane = t & 31;
    const int g = lane >> 4, ln = lane & 15;
    const int coBase = blockIdx.y * 128;
    const int pBase  = blockIdx.x * 128;
    const int mW = (wave >> 1) * 32, nW = (wave & 1) * 64;
    const int NS = KDIM / 32;

    // B geometry: thread owns 8 contiguous pixels at 2 kk values (loop-invariant)
    const int kk0 = t >> 4;                // 0..15 (second chunk at kk0+16)
    const int n0  = (t & 15) * 8;          // pixel group start within tile
    const int p0  = pBase + n0;
    const int py  = p0 / FEAT, px0 = p0 - py*FEAT;   // px0 % 8 == 0 -> rows don't split

    // A async geometry: 2 x b128 per thread per step
    const int am0 = t >> 2;                // m for j=0 (0..63)
    const int ach = t & 3;                 // 16B chunk within 64B k-row
    const unsigned sA0 = (unsigned)(uintptr_t)&sA[0][0];

    auto issueA = [&](int ks, int buf) {
        const int kg0 = ks * 32;
        #pragma unroll
        for (int j = 0; j < 2; ++j) {
            int m = am0 + j*64;
            unsigned loff = sA0 + (unsigned)(buf*128*40 + m*40 + ach*8) * 2u;
            async_ld_b128(loff, Wp + (size_t)(coBase+m)*KDIM + kg0 + ach*8);
        }
    };
    v8h bregv[2];
    auto gatherB = [&](int ks) {           // 2 coalesced b128 loads from shifted plane
        const int kg0 = ks * 32;
        const int khw = kg0 >> 10;
        const int ky = khw / 3, kx = khw - ky*3;
        const int c0 = kg0 & 1023;
        const _Float16* base = Xp3 + (size_t)kx*PLANE
                             + ((size_t)(c0 + kk0)*PH + (py + ky))*PW + px0;
        bregv[0] = *(const v8h*)base;
        bregv[1] = *(const v8h*)(base + (size_t)16*PH*PW);
    };

    v8f acc[2][4] = {};
    gatherB(0);
    issueA(0, 0);

    for (int ks = 0; ks < NS; ++ks) {
        #pragma unroll
        for (int j = 0; j < 2; ++j) {      // transpose-commit B regs to LDS
            union { v8h v; _Float16 e[8]; } u; u.v = bregv[j];
            const int kk = kk0 + j*16;
            #pragma unroll
            for (int e = 0; e < 8; ++e)
                sB[(n0 + e)*40 + kk] = u.e[e];
        }
        wait_async0();                     // A(ks) resident in LDS
        __syncthreads();
        if (ks + 1 < NS) {                 // overlap next-step staging with WMMA
            issueA(ks + 1, (ks + 1) & 1);
            gatherB(ks + 1);
        }
        const _Float16* bufA = &sA[ks & 1][0];

        v16h af[2], bf[4];
        #pragma unroll
        for (int mi = 0; mi < 2; ++mi) af[mi] = load_frag(bufA + (mW + mi*16 + ln)*40, g);
        #pragma unroll
        for (int ni = 0; ni < 4; ++ni) bf[ni] = load_frag(&sB[(nW + ni*16 + ln)*40], g);

        #pragma unroll
        for (int mi = 0; mi < 2; ++mi)
            #pragma unroll
            for (int ni = 0; ni < 4; ++ni)
                acc[mi][ni] = __builtin_amdgcn_wmma_f32_16x16x32_f16(
                    false, af[mi], false, bf[ni], (short)0, acc[mi][ni], false, false);
        __syncthreads();
    }

    // epilogue: C/D layout -> row = v + 8*g, col = ln ; store Hh pixel-major
    #pragma unroll
    for (int mi = 0; mi < 2; ++mi)
        #pragma unroll
        for (int ni = 0; ni < 4; ++ni)
            #pragma unroll
            for (int v = 0; v < 8; ++v) {
                int co = coBase + mW + mi*16 + v + 8*g;
                int pp = pBase + nW + ni*16 + ln;
                float val = acc[mi][ni][v] + brpn[co];
                val = val > 0.0f ? val : 0.0f;
                Hh[(size_t)pp*C_MID + co] = (_Float16)val;
            }
}

// ---------------- 1x1 convs: 64 x NPIX x 512 WMMA GEMM (fully async) -----
__global__ __launch_bounds__(128)
void gemm1x1_wmma(const _Float16* __restrict__ Hh,   // [NPIX][512]
                  const _Float16* __restrict__ Wcr,  // [64][512]
                  const float* __restrict__ bcls, const float* __restrict__ breg_,
                  float* __restrict__ cls_all,       // [N_ANCH][2]
                  float* __restrict__ reg_all)       // [N_ANCH][4]
{
    __shared__ __align__(16) _Float16 sA[64*40];
    __shared__ __align__(16) _Float16 sB[64*40];
    const int t = threadIdx.x;
    const int wave = t >> 5, lane = t & 31;
    const int g = lane >> 4, ln = lane & 15;
    const int pBase = blockIdx.x * 64;
    const int rm = t >> 2;                 // 0..31
    const int ch = t & 3;
    const unsigned sA0 = (unsigned)(uintptr_t)&sA[0];
    const unsigned sB0 = (unsigned)(uintptr_t)&sB[0];

    v8f acc[4] = {};
    for (int ks = 0; ks < C_MID/32; ++ks) {
        const int k0 = ks * 32;
        #pragma unroll
        for (int j = 0; j < 2; ++j) {
            int m = rm + j*32;
            unsigned off = (unsigned)(m*40 + ch*8) * 2u;
            async_ld_b128(sA0 + off, Wcr + (size_t)m*C_MID + k0 + ch*8);
            async_ld_b128(sB0 + off, Hh + (size_t)(pBase+m)*C_MID + k0 + ch*8);
        }
        wait_async0();
        __syncthreads();
        v16h af = load_frag(&sA[(wave*16 + ln)*40], g);
        #pragma unroll
        for (int ni = 0; ni < 4; ++ni) {
            v16h bf = load_frag(&sB[(ni*16 + ln)*40], g);
            acc[ni] = __builtin_amdgcn_wmma_f32_16x16x32_f16(
                false, af, false, bf, (short)0, acc[ni], false, false);
        }
        __syncthreads();
    }
    #pragma unroll
    for (int ni = 0; ni < 4; ++ni)
        #pragma unroll
        for (int v = 0; v < 8; ++v) {
            int m = wave*16 + v + 8*g;
            int p = pBase + ni*16 + ln;
            float val = acc[ni][v];
            if (m < 18) {
                int a = m >> 1, c = m & 1;
                cls_all[((size_t)p*KANCH + a)*2 + c] = val + bcls[m];
            } else if (m < 54) {
                int m2 = m - 18, a = m2 >> 2, j = m2 & 3;
                reg_all[((size_t)p*KANCH + a)*4 + j] = val + breg_[m2];
            }
        }
}

// ---------------- anchor decode + probs + sort keys ----------------
__global__ void decode_kernel(const float* __restrict__ anch,
                              const float* __restrict__ cls_all,
                              const float* __restrict__ reg_all,
                              float* __restrict__ proposals, float* __restrict__ bboxes,
                              float* __restrict__ probs, unsigned long long* __restrict__ keys)
{
    int i = blockIdx.x*blockDim.x + threadIdx.x;
    if (i < N_ANCH) {
        float a0 = anch[i*4+0], a1 = anch[i*4+1], a2 = anch[i*4+2], a3 = anch[i*4+3];
        float aw = a2 - a0 + 1.0f, ah = a3 - a1 + 1.0f;
        float ax = a0 + 0.5f*(aw - 1.0f), ay = a1 + 0.5f*(ah - 1.0f);
        float r0 = reg_all[i*4+0], r1 = reg_all[i*4+1], r2 = reg_all[i*4+2], r3 = reg_all[i*4+3];
        float px = ax + aw*r0, py = ay + ah*r1;
        float pw = aw*expf(r2), ph = ah*expf(r3);
        float x0 = px - 0.5f*(pw-1.0f), y0 = py - 0.5f*(ph-1.0f);
        float x1 = px + 0.5f*(pw-1.0f), y1 = py + 0.5f*(ph-1.0f);
        proposals[i*4+0] = x0; proposals[i*4+1] = y0;
        proposals[i*4+2] = x1; proposals[i*4+3] = y1;
        bboxes[i*4+0] = fminf(fmaxf(x0, 0.0f), IMG_MAX);
        bboxes[i*4+1] = fminf(fmaxf(y0, 0.0f), IMG_MAX);
        bboxes[i*4+2] = fminf(fmaxf(x1, 0.0f), IMG_MAX);
        bboxes[i*4+3] = fminf(fmaxf(y1, 0.0f), IMG_MAX);
        float c0 = cls_all[i*2+0], c1 = cls_all[i*2+1];
        float prob = 1.0f / (1.0f + expf(c0 - c1));
        probs[i] = prob;
        unsigned ub = __float_as_uint(prob);
        ub = (ub & 0x80000000u) ? ~ub : (ub | 0x80000000u);   // order-preserving flip
        unsigned k32 = ~ub;                                   // descending prob
        keys[i] = ((unsigned long long)k32 << 32) | (unsigned)i;  // stable ties
    } else if (i < N_SORT) {
        keys[i] = 0xFFFFFFFFFFFFFFFFull;                      // pad sorts last
    }
}

// ---------------- bitonic sort ----------------
__global__ void bitonic_step(unsigned long long* __restrict__ keys, int j, int k) {
    unsigned i = blockIdx.x*blockDim.x + threadIdx.x;
    unsigned ixj = i ^ (unsigned)j;
    if (ixj > i) {
        unsigned long long a = keys[i], b = keys[ixj];
        bool up = ((i & (unsigned)k) == 0);
        if (up ? (a > b) : (a < b)) { keys[i] = b; keys[ixj] = a; }
    }
}

__global__ void gather_top(const unsigned long long* __restrict__ keys,
                           const float* __restrict__ bboxes, const float* __restrict__ probs,
                           float* __restrict__ top_boxes, float* __restrict__ top_probs)
{
    int i = blockIdx.x*blockDim.x + threadIdx.x;
    if (i >= PRE_TOP) return;
    int idx = (int)(keys[i] & 0xFFFFFFFFull);
    top_boxes[i*4+0] = bboxes[idx*4+0];
    top_boxes[i*4+1] = bboxes[idx*4+1];
    top_boxes[i*4+2] = bboxes[idx*4+2];
    top_boxes[i*4+3] = bboxes[idx*4+3];
    top_probs[i] = probs[idx];
}

// ---------------- NMS: suppression bitmask ----------------
__global__ __launch_bounds__(64)
void nms_mask_kernel(const float* __restrict__ boxes, unsigned long long* __restrict__ mask) {
    __shared__ float4 sb[64];
    int colBase = blockIdx.x * 64;
    int row = blockIdx.y * 64 + threadIdx.x;
    int c = colBase + threadIdx.x;
    sb[threadIdx.x] = (c < PRE_TOP) ? ((const float4*)boxes)[c] : float4{0,0,0,0};
    __syncthreads();
    if (row >= PRE_TOP) return;
    float4 rb = ((const float4*)boxes)[row];
    float rArea = (rb.z - rb.x + 1.0f) * (rb.w - rb.y + 1.0f);
    unsigned long long bits = 0;
    for (int tt = 0; tt < 64; ++tt) {
        int col = colBase + tt;
        if (col < PRE_TOP && col > row) {
            float4 cb = sb[tt];
            float iw = fminf(rb.z, cb.z) - fmaxf(rb.x, cb.x) + 1.0f;
            float ih = fminf(rb.w, cb.w) - fmaxf(rb.y, cb.y) + 1.0f;
            iw = fmaxf(iw, 0.0f); ih = fmaxf(ih, 0.0f);
            float inter = iw * ih;
            float cArea = (cb.z - cb.x + 1.0f) * (cb.w - cb.y + 1.0f);
            float iou = inter / (rArea + cArea - inter);
            if (iou > NMS_THR_) bits |= (1ull << tt);
        }
    }
    mask[(size_t)row*NBLK64 + blockIdx.x] = bits;
}

// ---------------- NMS sequential reduce + scatter to output ----------------
__global__ __launch_bounds__(128)
void nms_reduce_kernel(const unsigned long long* __restrict__ mask,
                       const float* __restrict__ top_boxes, const float* __restrict__ top_probs,
                       float* __restrict__ filtered, float* __restrict__ probs_out)
{
    __shared__ unsigned long long remv[NBLK64];
    int t = threadIdx.x;
    if (t < NBLK64) remv[t] = 0ull;
    for (int i = t; i < POS_TOP*4; i += 128) filtered[i] = 0.0f;
    for (int i = t; i < POS_TOP;   i += 128) probs_out[i] = 0.0f;
    __syncthreads();
    int cnt = 0;   // only thread 0's copy is meaningful
    for (int i = 0; i < PRE_TOP; ++i) {
        bool removed = (remv[i >> 6] >> (i & 63)) & 1ull;
        __syncthreads();
        if (!removed) {
            if (t == 0) {
                if (cnt < POS_TOP) {
                    filtered[cnt*4+0] = top_boxes[i*4+0];
                    filtered[cnt*4+1] = top_boxes[i*4+1];
                    filtered[cnt*4+2] = top_boxes[i*4+2];
                    filtered[cnt*4+3] = top_boxes[i*4+3];
                    probs_out[cnt] = top_probs[i];
                }
                cnt++;
            }
            if (t < NBLK64) remv[t] |= mask[(size_t)i*NBLK64 + t];
        }
        __syncthreads();
    }
}

// ---------------- gather valid anchors into output head ----------------
__global__ void gather_valid(const int* __restrict__ vidx,
                             const float* __restrict__ proposals, const float* __restrict__ cls_all,
                             float* __restrict__ outP, float* __restrict__ outC, int nv)
{
    int i = blockIdx.x*blockDim.x + threadIdx.x;
    if (i >= nv) return;
    int v = vidx[i];
    outP[i*4+0] = proposals[v*4+0];
    outP[i*4+1] = proposals[v*4+1];
    outP[i*4+2] = proposals[v*4+2];
    outP[i*4+3] = proposals[v*4+3];
    outC[i*2+0] = cls_all[v*2+0];
    outC[i*2+1] = cls_all[v*2+1];
}

// ---------------- host orchestration ----------------
extern "C" void kernel_launch(void* const* d_in, const int* in_sizes, int n_in,
                              void* d_out, int out_size, void* d_ws, size_t ws_size,
                              hipStream_t stream) {
    const float* x      = (const float*)d_in[0];
    const float* w_rpn  = (const float*)d_in[1];
    const float* b_rpn  = (const float*)d_in[2];
    const float* w_cls  = (const float*)d_in[3];
    const float* b_cls  = (const float*)d_in[4];
    const float* w_reg  = (const float*)d_in[5];
    const float* b_reg  = (const float*)d_in[6];
    const float* anch   = (const float*)d_in[7];
    const int*   vidx   = (const int*)d_in[8];
    const int nv = in_sizes[8];

    char* ws = (char*)d_ws;
    size_t off = 0;
    auto alloc = [&](size_t bytes) -> void* {
        void* p = ws + off;
        off = (off + bytes + 255) & ~(size_t)255;
        return p;
    };
    _Float16* Xp3  = (_Float16*)alloc((size_t)3*PLANE*2);
    _Float16* Wp   = (_Float16*)alloc((size_t)C_MID*KDIM*2);
    _Float16* Wcr  = (_Float16*)alloc((size_t)64*C_MID*2);
    _Float16* Hh   = (_Float16*)alloc((size_t)NPIX*C_MID*2);
    float* cls_all = (float*)alloc((size_t)N_ANCH*2*4);
    float* reg_all = (float*)alloc((size_t)N_ANCH*4*4);
    float* props   = (float*)alloc((size_t)N_ANCH*4*4);
    float* bboxes  = (float*)alloc((size_t)N_ANCH*4*4);
    float* probs   = (float*)alloc((size_t)N_ANCH*4);
    unsigned long long* keys = (unsigned long long*)alloc((size_t)N_SORT*8);
    float* top_boxes = (float*)alloc((size_t)PRE_TOP*4*4);
    float* top_probs = (float*)alloc((size_t)PRE_TOP*4);
    unsigned long long* nmask = (unsigned long long*)alloc((size_t)PRE_TOP*NBLK64*8);

    float* out   = (float*)d_out;
    float* outP  = out;                            // nv*4 proposals[valid_idx]
    float* outC  = out + (size_t)nv*4;             // nv*2 cls[valid_idx]
    float* outF  = out + (size_t)nv*6;             // 300*4 filtered
    float* outPr = out + (size_t)nv*6 + POS_TOP*4; // 300 probs_out

    {
        int n = (int)(3*PLANE);
        pack_x_shift3<<<(n+255)/256, 256, 0, stream>>>(x, Xp3, n);
        pack_w_rpn<<<(C_MID*KDIM+255)/256, 256, 0, stream>>>(w_rpn, Wp);
        pack_wcr<<<(64*C_MID+255)/256, 256, 0, stream>>>(w_cls, w_reg, Wcr);
    }
    conv3x3_wmma<<<dim3(NPIX/128, C_MID/128), 256, 0, stream>>>(Xp3, Wp, b_rpn, Hh);
    gemm1x1_wmma<<<NPIX/64, 128, 0, stream>>>(Hh, Wcr, b_cls, b_reg, cls_all, reg_all);
    decode_kernel<<<N_SORT/256, 256, 0, stream>>>(anch, cls_all, reg_all,
                                                  props, bboxes, probs, keys);
    for (int k = 2; k <= N_SORT; k <<= 1)
        for (int j = k >> 1; j >= 1; j >>= 1)
            bitonic_step<<<N_SORT/256, 256, 0, stream>>>(keys, j, k);
    gather_top<<<(PRE_TOP+255)/256, 256, 0, stream>>>(keys, bboxes, probs, top_boxes, top_probs);
    nms_mask_kernel<<<dim3(NBLK64, NBLK64), 64, 0, stream>>>(top_boxes, nmask);
    nms_reduce_kernel<<<1, 128, 0, stream>>>(nmask, top_boxes, top_probs, outF, outPr);
    gather_valid<<<(nv+255)/256, 256, 0, stream>>>(vidx, props, cls_all, outP, outC, nv);
    (void)n_in; (void)out_size; (void)ws_size;
}
